// MultiHeadAttention_88115549044991
// MI455X (gfx1250) — matrix operand order repaired and checked by reading, not compiled, not run on hardware
//
#include <hip/hip_runtime.h>

// Multi-head attention for MI455X (gfx1250), wave32 + v_wmma_f32_16x16x32_f16.
// B=8, S=1024, DM=512, H=8, D=64.  Outputs: z [B,S,DM] fp32 then attn [B,H,S,S] fp32.

#define BATCH  8
#define SEQ    1024
#define DMODEL 512
#define NHEAD  8
#define HDIM   64
#define FDIM   (NHEAD * HDIM)   // 512

typedef __attribute__((ext_vector_type(16))) _Float16 v16h;
typedef __attribute__((ext_vector_type(8)))  float    v8f;

union FragH {
    v16h     v;
    uint32_t u[8];
    _Float16 h[16];
};

// ---------------------------------------------------------------------------
// Fragment loaders (wave32 WMMA layouts; C/D per CDNA5 ISA 7.12.2, A/B in this code's contiguous k order)
// A 16x32 f16:   lane L holds row M=L%16; K(v,p) = (L>=16?16:0) + v*2 + p
// B 32x16 f16 from a transposed (N-major) buffer bt[n][k]:
//                lane L holds col N=L%16; K(v,p) = (L>=16?16:0) + v*2 + p  (contiguous -> b128)
// C/D 16x16 f32: lane L holds col N=L%16; VGPR r holds row M = r + (L>=16?8:0)
// ---------------------------------------------------------------------------
__device__ __forceinline__ v16h load_a_f16(const _Float16* base, int ld, int lane)
{
    const int row = lane & 15, hi = lane >> 4;
    const uint32_t* q = reinterpret_cast<const uint32_t*>(base + (size_t)row * ld);
    FragH f;
#pragma unroll
    for (int v = 0; v < 8; ++v) {
        const int k = hi * 16 + v * 2;
        f.u[v] = q[k >> 1];
    }
    return f.v;
}

__device__ __forceinline__ v16h load_b_t(const _Float16* bt, int ld, int lane)
{
    const int col = lane & 15, hi = lane >> 4;
    const uint32_t* q = reinterpret_cast<const uint32_t*>(bt + (size_t)col * ld + hi * 16);
    FragH f;
#pragma unroll
    for (int v = 0; v < 8; ++v) f.u[v] = q[v];   // 32B contiguous -> global_load_b128 x2
    return f.v;
}

// A fragment from an LDS-staged 16x32 fp32 tile (row stride 32 floats)
__device__ __forceinline__ v16h frag_a_lds_f32(const float* lb, int lane)
{
    const int row = lane & 15, hi = lane >> 4;
    const float* p = lb + row * 32;
    FragH f;
#pragma unroll
    for (int v = 0; v < 8; ++v) {
        const int k = hi * 16 + v * 2;
        f.h[2 * v]     = (_Float16)p[k];
        f.h[2 * v + 1] = (_Float16)p[k + 1];
    }
    return f.v;
}

// ---------------------------------------------------------------------------
// Conversion / transpose kernels
// ---------------------------------------------------------------------------
__global__ void cvt_f32_f16(const float* __restrict__ src, _Float16* __restrict__ dst, int n)
{
    int i = blockIdx.x * blockDim.x + threadIdx.x;
    if (i < n) dst[i] = (_Float16)src[i];
}

// src: [R][C] row-major f32  ->  dst: [C][R] row-major f16
__global__ void transpose_cvt(const float* __restrict__ src, _Float16* __restrict__ dst,
                              int R, int C)
{
    int i = blockIdx.x * blockDim.x + threadIdx.x;
    if (i < R * C) {
        int r = i / C, c = i - r * C;
        dst[(size_t)c * R + r] = (_Float16)src[i];
    }
}

// ---------------------------------------------------------------------------
// QKV projection:  out[b,h] = x[b] (SxDM) * W[h] (DMxD), W given transposed [h][d][m].
// Each wave computes a 16x64 tile: A fragment + all 4 B fragments preloaded into
// distinct registers so the WMMA chain overlaps outstanding loads (progressive
// s_wait_loadcnt instead of full drains).
// vTranspose=0: row-major f16 out [b,h,s,d];  =1: out is V^T [b,h,d,s] (16B packed stores)
// grid.x = (B*H) * (S/128) = 512, block = 256 (8 waves, 16 rows each)
// ---------------------------------------------------------------------------
__global__ void qkv_gemm(const _Float16* __restrict__ xh,
                         const _Float16* __restrict__ wT,
                         _Float16* __restrict__ out,
                         int vTranspose)
{
    const int lane = threadIdx.x & 31;
    const int w    = threadIdx.x >> 5;
    int bx = blockIdx.x;
    const int sb = bx & 7;  bx >>= 3;
    const int bh = bx;                 // 0..63
    const int b  = bh >> 3;
    const int h  = bh & 7;
    const int s0 = sb * 128 + w * 16;

    const _Float16* abase = xh + ((size_t)(b * SEQ + s0)) * DMODEL;
    const _Float16* bbase = wT + ((size_t)h * HDIM) * DMODEL;

    v8f acc[4] = {};
#pragma unroll 2
    for (int k = 0; k < DMODEL; k += 32) {
        v16h a = load_a_f16(abase + k, DMODEL, lane);
        v16h bf[4];
#pragma unroll
        for (int j = 0; j < 4; ++j)
            bf[j] = load_b_t(bbase + (size_t)j * 16 * DMODEL + k, DMODEL, lane);
#pragma unroll
        for (int j = 0; j < 4; ++j)
            acc[j] = __builtin_amdgcn_wmma_f32_16x16x32_f16(false, a, false, bf[j],
                                                            (short)0, acc[j], false, false);
    }

    const int col = lane & 15, hi = lane >> 4;
    if (vTranspose) {
#pragma unroll
        for (int j = 0; j < 4; ++j) {
            union { _Float16 h8[8]; uint4 q; } pk;
#pragma unroll
            for (int r = 0; r < 8; ++r) pk.h8[r] = (_Float16)acc[j][r];  // M=r+hi*8 -> contiguous s
            _Float16* dst = out + ((size_t)(bh * HDIM + j * 16 + col)) * SEQ + s0 + hi * 8;
            *reinterpret_cast<uint4*>(dst) = pk.q;                        // 16B packed store
        }
    } else {
        _Float16* dst = out + ((size_t)(bh * SEQ + s0)) * HDIM + col;
#pragma unroll
        for (int j = 0; j < 4; ++j)
#pragma unroll
            for (int r = 0; r < 8; ++r)
                dst[(size_t)(r + hi * 8) * HDIM + j * 16] = (_Float16)acc[j][r];
    }
}

// ---------------------------------------------------------------------------
// scores[b,h,s,t] = (Q . K) / 8.  Wave tile 16(s) x 64(t), 4 accumulators.
// grid.x = 64 * (S/128) * (S/64) = 8192, block = 256
// ---------------------------------------------------------------------------
__global__ void qk_gemm(const _Float16* __restrict__ Qh,
                        const _Float16* __restrict__ Kh,
                        float* __restrict__ attn)
{
    const int lane = threadIdx.x & 31;
    const int w    = threadIdx.x >> 5;
    int bx = blockIdx.x;
    const int tb = bx & 15; bx >>= 4;
    const int sb = bx & 7;  bx >>= 3;
    const int bh = bx;
    const int s0 = sb * 128 + w * 16;
    const int t0 = tb * 64;

    const _Float16* abase = Qh + ((size_t)(bh * SEQ + s0)) * HDIM;
    const _Float16* bbase = Kh + ((size_t)(bh * SEQ + t0)) * HDIM;   // K row-major == B^T

    v8f acc[4] = {};
#pragma unroll
    for (int k = 0; k < HDIM; k += 32) {
        v16h a = load_a_f16(abase + k, HDIM, lane);
        v16h bf[4];
#pragma unroll
        for (int j = 0; j < 4; ++j)
            bf[j] = load_b_t(bbase + (size_t)j * 16 * HDIM + k, HDIM, lane);
#pragma unroll
        for (int j = 0; j < 4; ++j)
            acc[j] = __builtin_amdgcn_wmma_f32_16x16x32_f16(false, a, false, bf[j],
                                                            (short)0, acc[j], false, false);
    }
    const int col = lane & 15, hi = lane >> 4;
    float* dst = attn + ((size_t)bh * SEQ + s0) * SEQ + t0 + col;
#pragma unroll
    for (int j = 0; j < 4; ++j)
#pragma unroll
        for (int r = 0; r < 8; ++r)
            dst[(size_t)(r + hi * 8) * SEQ + j * 16] = acc[j][r] * 0.125f;
}

// ---------------------------------------------------------------------------
// In-place row softmax over S=1024.  One 256-thread block per row, float4 per thread.
// ---------------------------------------------------------------------------
__global__ void softmax_rows(float* __restrict__ attn)
{
    __shared__ float red[256];
    const int tid = threadIdx.x;
    float* row = attn + (size_t)blockIdx.x * SEQ;
    float4 xv = reinterpret_cast<float4*>(row)[tid];

    float m = fmaxf(fmaxf(xv.x, xv.y), fmaxf(xv.z, xv.w));
    red[tid] = m;
    __syncthreads();
#pragma unroll
    for (int s = 128; s > 0; s >>= 1) {
        if (tid < s) red[tid] = fmaxf(red[tid], red[tid + s]);
        __syncthreads();
    }
    m = red[0];
    __syncthreads();

    xv.x = __expf(xv.x - m);
    xv.y = __expf(xv.y - m);
    xv.z = __expf(xv.z - m);
    xv.w = __expf(xv.w - m);
    red[tid] = xv.x + xv.y + xv.z + xv.w;
    __syncthreads();
#pragma unroll
    for (int s = 128; s > 0; s >>= 1) {
        if (tid < s) red[tid] += red[tid + s];
        __syncthreads();
    }
    const float inv = 1.0f / red[0];

    xv.x *= inv; xv.y *= inv; xv.z *= inv; xv.w *= inv;
    reinterpret_cast<float4*>(row)[tid] = xv;
}

// ---------------------------------------------------------------------------
// heads = attn (SxS fp32) * V (SxD via V^T).  Wave tile 16(s) x 64(d), 4 accumulators.
// attn A-tiles are double-buffer prefetched into LDS with ASYNC loads
// (global_load_async_to_lds_b128 + s_wait_asynccnt), then converted f32->f16.
// Each attn element is read exactly once from memory.
// grid.x = 64 * (S/128) = 512, block = 256.  LDS: 8 waves * 2 bufs * 2KB = 32KB.
// ---------------------------------------------------------------------------
__global__ void av_gemm(const float* __restrict__ attn,
                        const _Float16* __restrict__ VTh,
                        _Float16* __restrict__ concat)
{
    __shared__ float smem[8 * 2 * 512];   // [wave][buf][16 rows x 32 f32]

    const int lane = threadIdx.x & 31;
    const int w    = threadIdx.x >> 5;
    int bx = blockIdx.x;
    const int sb = bx & 7;  bx >>= 3;
    const int bh = bx;
    const int b  = bh >> 3;
    const int h  = bh & 7;
    const int s0 = sb * 128 + w * 16;

    const float*    abase = attn + ((size_t)bh * SEQ + s0) * SEQ;
    const _Float16* bbase = VTh + ((size_t)bh * HDIM) * SEQ;

    const float* lwave = &smem[w * 1024];
    const uint32_t lds_base = (uint32_t)(uintptr_t)lwave;   // LDS offset = low 32 bits
    const int arow = lane >> 3;   // 0..3
    const int aseg = lane & 7;    // 0..7 (16B segments of a 128B row)

    // Stage a 16x32 fp32 attn tile into LDS buffer `buf` (4 async b128 ops, 512B each)
    auto stage = [&](int buf, int k) {
#pragma unroll
        for (int jj = 0; jj < 4; ++jj) {
            const int row = arow + 4 * jj;
            uint32_t dstoff = lds_base + (uint32_t)buf * 2048u
                            + (uint32_t)(row * 128 + aseg * 16);
            const float* src = abase + (size_t)row * SEQ + k + aseg * 4;
            asm volatile("global_load_async_to_lds_b128 %0, %1, off"
                         :: "v"(dstoff), "v"(src)
                         : "memory");
        }
    };

    stage(0, 0);
    v8f acc[4] = {};
    int buf = 0;
    for (int k = 0; k < SEQ; k += 32) {
        if (k + 32 < SEQ) {
            stage(buf ^ 1, k + 32);                          // prefetch next tile
            asm volatile("s_wait_asynccnt 4" ::: "memory");  // current tile complete
        } else {
            asm volatile("s_wait_asynccnt 0" ::: "memory");
        }
        v16h a = frag_a_lds_f32(lwave + buf * 512, lane);
        v16h bf[4];
#pragma unroll
        for (int j = 0; j < 4; ++j)
            bf[j] = load_b_t(bbase + (size_t)j * 16 * SEQ + k, SEQ, lane);
#pragma unroll
        for (int j = 0; j < 4; ++j)
            acc[j] = __builtin_amdgcn_wmma_f32_16x16x32_f16(false, a, false, bf[j],
                                                            (short)0, acc[j], false, false);
        buf ^= 1;
    }

    const int col = lane & 15, hi = lane >> 4;
    _Float16* dst = concat + ((size_t)(b * SEQ + s0)) * FDIM + h * HDIM + col;
#pragma unroll
    for (int j = 0; j < 4; ++j)
#pragma unroll
        for (int r = 0; r < 8; ++r)
            dst[(size_t)(r + hi * 8) * FDIM + j * 16] = (_Float16)acc[j][r];
}

// ---------------------------------------------------------------------------
// z = concat (B*S x 512) * Wo (512x512), Wo given transposed [m][f].  fp32 out.
// Wave tile 16(r) x 64(n), 4 accumulators.  grid.x = (8192/128) * (512/64) = 512
// ---------------------------------------------------------------------------
__global__ void out_proj(const _Float16* __restrict__ concat,
                         const _Float16* __restrict__ woT,
                         float* __restrict__ z)
{
    const int lane = threadIdx.x & 31;
    const int w    = threadIdx.x >> 5;
    int bx = blockIdx.x;
    const int nb = bx & 7;  bx >>= 3;
    const int rb = bx;
    const int r0 = rb * 128 + w * 16;
    const int n0 = nb * 64;

    const _Float16* abase = concat + (size_t)r0 * FDIM;
    const _Float16* bbase = woT + (size_t)n0 * FDIM;

    v8f acc[4] = {};
#pragma unroll 2
    for (int k = 0; k < FDIM; k += 32) {
        v16h a = load_a_f16(abase + k, FDIM, lane);
        v16h bf[4];
#pragma unroll
        for (int j = 0; j < 4; ++j)
            bf[j] = load_b_t(bbase + (size_t)j * 16 * FDIM + k, FDIM, lane);
#pragma unroll
        for (int j = 0; j < 4; ++j)
            acc[j] = __builtin_amdgcn_wmma_f32_16x16x32_f16(false, a, false, bf[j],
                                                            (short)0, acc[j], false, false);
    }
    const int col = lane & 15, hi = lane >> 4;
    float* dst = z + (size_t)r0 * DMODEL + n0 + col;
#pragma unroll
    for (int j = 0; j < 4; ++j)
#pragma unroll
        for (int r = 0; r < 8; ++r)
            dst[(size_t)(r + hi * 8) * DMODEL + j * 16] = acc[j][r];
}

// ---------------------------------------------------------------------------
// Host driver
// ---------------------------------------------------------------------------
extern "C" void kernel_launch(void* const* d_in, const int* in_sizes, int n_in,
                              void* d_out, int out_size, void* d_ws, size_t ws_size,
                              hipStream_t stream)
{
    (void)in_sizes; (void)n_in; (void)out_size; (void)ws_size;

    const float* x  = (const float*)d_in[0];
    const float* Wq = (const float*)d_in[1];
    const float* Wk = (const float*)d_in[2];
    const float* Wv = (const float*)d_in[3];
    const float* Wo = (const float*)d_in[4];

    float* zout = (float*)d_out;                               // [B,S,DM]
    float* attn = zout + (size_t)BATCH * SEQ * DMODEL;         // [B,H,S,S]

    char*  ws  = (char*)d_ws;
    size_t off = 0;
    auto carve = [&](size_t elems) -> _Float16* {
        _Float16* p = (_Float16*)(ws + off);
        off += ((elems * sizeof(_Float16)) + 255) & ~(size_t)255;
        return p;
    };
    _Float16* xh   = carve((size_t)BATCH * SEQ * DMODEL);
    _Float16* wqT  = carve((size_t)NHEAD * DMODEL * HDIM);
    _Float16* wkT  = carve((size_t)NHEAD * DMODEL * HDIM);
    _Float16* wvT  = carve((size_t)NHEAD * DMODEL * HDIM);
    _Float16* woT  = carve((size_t)FDIM * DMODEL);
    _Float16* Qh   = carve((size_t)BATCH * NHEAD * SEQ * HDIM);
    _Float16* Kh   = carve((size_t)BATCH * NHEAD * SEQ * HDIM);
    _Float16* VTh  = carve((size_t)BATCH * NHEAD * SEQ * HDIM);
    _Float16* conc = carve((size_t)BATCH * SEQ * FDIM);

    // fp32 -> f16 conversions (+ weight transposes for contiguous-K B fragments)
    {
        int n = BATCH * SEQ * DMODEL;
        cvt_f32_f16<<<(n + 255) / 256, 256, 0, stream>>>(x, xh, n);
    }
    {
        int n = DMODEL * HDIM;
        int g = (n + 255) / 256;
        for (int h = 0; h < NHEAD; ++h) {
            transpose_cvt<<<g, 256, 0, stream>>>(Wq + (size_t)h * n, wqT + (size_t)h * n, DMODEL, HDIM);
            transpose_cvt<<<g, 256, 0, stream>>>(Wk + (size_t)h * n, wkT + (size_t)h * n, DMODEL, HDIM);
            transpose_cvt<<<g, 256, 0, stream>>>(Wv + (size_t)h * n, wvT + (size_t)h * n, DMODEL, HDIM);
        }
    }
    {
        int n = FDIM * DMODEL;
        transpose_cvt<<<(n + 255) / 256, 256, 0, stream>>>(Wo, woT, FDIM, DMODEL);
    }

    const dim3 blk(256);
    qkv_gemm<<<512, blk, 0, stream>>>(xh, wqT, Qh, 0);
    qkv_gemm<<<512, blk, 0, stream>>>(xh, wkT, Kh, 0);
    qkv_gemm<<<512, blk, 0, stream>>>(xh, wvT, VTh, 1);

    qk_gemm<<<8192, blk, 0, stream>>>(Qh, Kh, attn);
    softmax_rows<<<BATCH * NHEAD * SEQ, blk, 0, stream>>>(attn);
    av_gemm<<<512, blk, 0, stream>>>(attn, VTh, conc);
    out_proj<<<512, blk, 0, stream>>>(conc, woT, zout);
}
